// MultiHeadSelfAttention2DRoPE_61357902791025
// MI455X (gfx1250) — compile-verified, matrix operand order checked by reading, not run on hardware
//
#include <hip/hip_runtime.h>

typedef __attribute__((ext_vector_type(16))) _Float16 v16h;
typedef __attribute__((ext_vector_type(8)))  _Float16 v8h;
typedef __attribute__((ext_vector_type(8)))  float    v8f;
typedef __attribute__((ext_vector_type(4)))  float    v4f;

#define B_    8
#define L_    1024
#define DIM_  1024
#define NH    16
#define DH    64
#define N3    3072
#define PI_F  3.14159265358979323846f

// ---------------------------------------------------------------------------
// WMMA fragment helpers (CDNA5 16x16x32 f16, wave32)
// ---------------------------------------------------------------------------
__device__ __forceinline__ v16h frag_f16(const _Float16* base, int stride, int lane) {
  const int row = lane & 15;
  const int lo  = (lane & 16) ? 8 : 0;
  const _Float16* p = base + row * stride + lo;
  v8h a = *(const v8h*)(p);
  v8h b = *(const v8h*)(p + 16);
  return __builtin_shufflevector(a, b, 0,1,2,3,4,5,6,7,8,9,10,11,12,13,14,15);
}

__device__ __forceinline__ v16h frag_f32(const float* base, int stride, int lane) {
  const int row = lane & 15;
  const int lo  = (lane & 16) ? 8 : 0;
  const float* p = base + row * stride + lo;
  v16h r;
#pragma unroll
  for (int i = 0; i < 8; ++i) { r[i] = (_Float16)p[i]; r[i + 8] = (_Float16)p[16 + i]; }
  return r;
}

__device__ __forceinline__ v8f wmma16(v16h a, v16h b, v8f c) {
  return __builtin_amdgcn_wmma_f32_16x16x32_f16(false, a, false, b, (short)0, c, false, false);
}

// CDNA5 async copy: 16B per lane, global -> LDS, tracked by ASYNCcnt
__device__ __forceinline__ void async_ld16(unsigned lds_byte_off, const void* gptr) {
  asm volatile("global_load_async_to_lds_b128 %0, %1, off"
               :: "v"(lds_byte_off), "v"(gptr) : "memory");
}
__device__ __forceinline__ void wait_async_le0() {
  asm volatile("s_wait_asynccnt 0x0" ::: "memory");
}
__device__ __forceinline__ void wait_async_le8() {
  asm volatile("s_wait_asynccnt 0x8" ::: "memory");
}

// ---------------------------------------------------------------------------
// Kernel 1: QKV = x @ w_qkv with fused 2D axial RoPE; q pre-scaled by 1/64.
// BM=64, BN=256, BK=32; 8 waves = 2(M) x 4(N); software-pipelined B staging.
// ---------------------------------------------------------------------------
__global__ __launch_bounds__(256) void qkv_rope_kernel(
    const float* __restrict__ x, const float* __restrict__ w,
    _Float16* __restrict__ qw, _Float16* __restrict__ kw, _Float16* __restrict__ vw) {
  __shared__ _Float16 Bs[256 * 40];  // [n][k], stride 40 halves (80B, 16B aligned)
  const int lane = threadIdx.x & 31;
  const int wid  = threadIdx.x >> 5;
  const int wm = wid & 1, wn = wid >> 1;
  const int bn = blockIdx.x, bm = blockIdx.y;
  const int row0 = bm * 64 + wm * 32;

  v8f z = {0.f,0.f,0.f,0.f,0.f,0.f,0.f,0.f};
  v8f acc[2][4];
#pragma unroll
  for (int ms = 0; ms < 2; ++ms)
#pragma unroll
    for (int nt = 0; nt < 4; ++nt) acc[ms][nt] = z;

  v4f tmp[8];                                    // register double-buffer for B
#pragma unroll
  for (int i = 0; i < 8; ++i) {                  // prologue: tile kb=0
    int f4 = threadIdx.x + 256 * i;
    int k = f4 >> 6, n4 = f4 & 63;
    tmp[i] = *(const v4f*)&w[(size_t)k * N3 + bn * 256 + n4 * 4];
  }

  for (int kb = 0; kb < DIM_; kb += 32) {
    __syncthreads();                             // previous tile's reads done
#pragma unroll
    for (int i = 0; i < 8; ++i) {                // regs -> LDS (cvt + transpose)
      int f4 = threadIdx.x + 256 * i;
      int k = f4 >> 6, n = (f4 & 63) * 4;
      Bs[(n + 0) * 40 + k] = (_Float16)tmp[i][0];
      Bs[(n + 1) * 40 + k] = (_Float16)tmp[i][1];
      Bs[(n + 2) * 40 + k] = (_Float16)tmp[i][2];
      Bs[(n + 3) * 40 + k] = (_Float16)tmp[i][3];
    }
    __syncthreads();
    if (kb + 32 < DIM_) {                        // issue next tile early: overlaps WMMA
#pragma unroll
      for (int i = 0; i < 8; ++i) {
        int f4 = threadIdx.x + 256 * i;
        int k = f4 >> 6, n4 = f4 & 63;
        tmp[i] = *(const v4f*)&w[(size_t)(kb + 32 + k) * N3 + bn * 256 + n4 * 4];
      }
    }
    v16h a0 = frag_f32(x + (size_t)row0 * DIM_ + kb, DIM_, lane);
    v16h a1 = frag_f32(x + (size_t)(row0 + 16) * DIM_ + kb, DIM_, lane);
    v16h bf[4];
#pragma unroll
    for (int nt = 0; nt < 4; ++nt)               // batch ds loads, stagger waits
      bf[nt] = frag_f16(Bs + (wn * 64 + nt * 16) * 40, 40, lane);
#pragma unroll
    for (int nt = 0; nt < 4; ++nt) {
      acc[0][nt] = wmma16(a0, bf[nt], acc[0][nt]);
      acc[1][nt] = wmma16(a1, bf[nt], acc[1][nt]);
    }
  }

  // Epilogue: RoPE + scatter. C layout: lane holds (M = r + hi, N = lane&15).
  const int hi = (lane & 16) ? 8 : 0;
#pragma unroll
  for (int ms = 0; ms < 2; ++ms) {
#pragma unroll
    for (int nt = 0; nt < 4; ++nt) {
      int n = bn * 256 + wn * 64 + nt * 16 + (lane & 15);
      int seg = n >> 10;                 // 0=q 1=k 2=v (uniform per tile)
      int c = n & 1023;
      int h = c >> 6, d = c & 63;
#pragma unroll
      for (int r = 0; r < 8; ++r) {
        int m = row0 + ms * 16 + r + hi;
        int b = m >> 10, l = m & 1023;
        float val = acc[ms][nt][r];
        float partner = __shfl_xor(val, 1, 32);  // rotate-half pair = adjacent lane
        if (seg < 2) {
          int dp = d & 62;
          int j  = ((dp >= 32) ? (dp - 32) : dp) >> 1;
          float fj  = (1.0f + (127.0f / 15.0f) * (float)j) * PI_F;
          float pos = (dp >= 32) ? (-1.0f + (2.0f / 31.0f) * (float)(l & 31))
                                 : (-1.0f + (2.0f / 31.0f) * (float)(l >> 5));
          float th = pos * fj;
          float cv = __cosf(th), sv = __sinf(th);
          float res = (d & 1) ? (val * cv + partner * sv)
                              : (val * cv - partner * sv);
          size_t o = ((size_t)(b * NH + h) * L_ + l) * DH + d;
          if (seg == 0) qw[o] = (_Float16)(res * (1.0f / 64.0f));
          else          kw[o] = (_Float16)res;
        } else {
          vw[((size_t)(b * NH + h) * DH + d) * L_ + l] = (_Float16)val;  // transposed
        }
      }
    }
  }
}

// ---------------------------------------------------------------------------
// Kernel 2: flash attention per (b,h,64-query slab).
// Double-buffered async-to-LDS K/V staging pipelined on ASYNCcnt.
// ---------------------------------------------------------------------------
__global__ __launch_bounds__(128) void attn_kernel(
    const _Float16* __restrict__ qw, const _Float16* __restrict__ kw,
    const _Float16* __restrict__ vw, _Float16* __restrict__ ao) {
  __shared__ _Float16 Qs[64 * 72];       // [m][d]
  __shared__ _Float16 Ks[2][64 * 72];    // [key][d], double-buffered
  __shared__ _Float16 Vs[2][64 * 72];    // [d][key], double-buffered
  __shared__ _Float16 Ps[64 * 72];       // per-wave 16-row strips
  const int lane = threadIdx.x & 31;
  const int wid  = threadIdx.x >> 5;
  const int qb = blockIdx.x, h = blockIdx.y, b = blockIdx.z;
  const int bh = b * NH + h;
  const _Float16* qg = qw + (size_t)bh * L_ * DH + (size_t)qb * 64 * DH;
  const _Float16* kg = kw + (size_t)bh * L_ * DH;
  const _Float16* vg = vw + (size_t)bh * DH * L_;

  const unsigned qs0 = (unsigned)(uintptr_t)(&Qs[0]);
  const unsigned ks0 = (unsigned)(uintptr_t)(&Ks[0][0]);
  const unsigned vs0 = (unsigned)(uintptr_t)(&Vs[0][0]);
  const unsigned bufstep = 64 * 72 * 2;  // bytes per buffer

  // per-thread slice used for all tile copies: 4 rows of 16B
  const int off0 = threadIdx.x * 8;      // covers 1024 halves per c-step
  // stage Q (4 async) then K/V tile 0 (8 async)
#pragma unroll
  for (int c = 0; c < 4; ++c) {
    int off = off0 + 128 * 8 * c;
    int m = off >> 6, d0 = off & 63;
    async_ld16(qs0 + (unsigned)(m * 72 + d0) * 2, qg + m * DH + d0);
  }
#pragma unroll
  for (int c = 0; c < 4; ++c) {
    int off = off0 + 128 * 8 * c;
    int row = off >> 6, d0 = off & 63;
    async_ld16(ks0 + (unsigned)(row * 72 + d0) * 2, kg + (size_t)row * DH + d0);
    async_ld16(vs0 + (unsigned)(row * 72 + d0) * 2, vg + (size_t)row * L_ + d0);
  }
  wait_async_le8();                      // Q (oldest 4) complete
  __syncthreads();

  // Q fragments are kt-invariant: hoist out of the loop
  v16h qf[2];
#pragma unroll
  for (int kk = 0; kk < 2; ++kk)
    qf[kk] = frag_f16(Qs + wid * 16 * 72 + kk * 32, 72, lane);

  v8f z = {0.f,0.f,0.f,0.f,0.f,0.f,0.f,0.f};
  v8f o[4];
#pragma unroll
  for (int dt = 0; dt < 4; ++dt) o[dt] = z;
  float mi[8], li[8];
#pragma unroll
  for (int r = 0; r < 8; ++r) { mi[r] = -1e30f; li[r] = 0.0f; }
  const int hi = (lane & 16) ? 8 : 0;

  for (int kt = 0; kt < 16; ++kt) {
    const int cur = kt & 1;
    if (kt + 1 < 16) {                   // issue next tile; overlaps compute below
      const int nxt = (kt + 1) & 1;
#pragma unroll
      for (int c = 0; c < 4; ++c) {
        int off = off0 + 128 * 8 * c;
        int row = off >> 6, d0 = off & 63;
        async_ld16(ks0 + nxt * bufstep + (unsigned)(row * 72 + d0) * 2,
                   kg + (size_t)((kt + 1) * 64 + row) * DH + d0);
        async_ld16(vs0 + nxt * bufstep + (unsigned)(row * 72 + d0) * 2,
                   vg + (size_t)row * L_ + (kt + 1) * 64 + d0);
      }
      wait_async_le8();                  // in-order: tile kt fully landed
    } else {
      wait_async_le0();
    }
    __syncthreads();                     // all waves' tile-kt parts visible

    v8f s[4];
#pragma unroll
    for (int nt = 0; nt < 4; ++nt) s[nt] = z;
#pragma unroll
    for (int kk = 0; kk < 2; ++kk) {
      v16h bf[4];
#pragma unroll
      for (int nt = 0; nt < 4; ++nt)
        bf[nt] = frag_f16(&Ks[cur][0] + nt * 16 * 72 + kk * 32, 72, lane);
#pragma unroll
      for (int nt = 0; nt < 4; ++nt)
        s[nt] = wmma16(qf[kk], bf[nt], s[nt]);
    }

    // Online softmax: row reductions inside 16-lane halves (xor 1,2,4,8)
    _Float16* pw = Ps + wid * 16 * 72;
#pragma unroll
    for (int r = 0; r < 8; ++r) {
      float mx = s[0][r];
#pragma unroll
      for (int nt = 1; nt < 4; ++nt) mx = fmaxf(mx, s[nt][r]);
      mx = fmaxf(mx, __shfl_xor(mx, 1, 32));
      mx = fmaxf(mx, __shfl_xor(mx, 2, 32));
      mx = fmaxf(mx, __shfl_xor(mx, 4, 32));
      mx = fmaxf(mx, __shfl_xor(mx, 8, 32));
      float mnew  = fmaxf(mi[r], mx);
      float alpha = __expf(mi[r] - mnew);
      mi[r] = mnew;
      float sum = 0.0f;
#pragma unroll
      for (int nt = 0; nt < 4; ++nt) {
        float p = __expf(s[nt][r] - mnew);
        s[nt][r] = p;
        sum += p;
      }
      sum += __shfl_xor(sum, 1, 32);
      sum += __shfl_xor(sum, 2, 32);
      sum += __shfl_xor(sum, 4, 32);
      sum += __shfl_xor(sum, 8, 32);
      li[r] = li[r] * alpha + sum;
#pragma unroll
      for (int dt = 0; dt < 4; ++dt) o[dt][r] *= alpha;
    }

    // Spill P (per-wave private LDS strip; LDS in-order within a wave)
#pragma unroll
    for (int nt = 0; nt < 4; ++nt)
#pragma unroll
      for (int r = 0; r < 8; ++r)
        pw[(r + hi) * 72 + nt * 16 + (lane & 15)] = (_Float16)s[nt][r];

    // O += P @ V
#pragma unroll
    for (int kk = 0; kk < 2; ++kk) {
      v16h a = frag_f16(pw + kk * 32, 72, lane);
      v16h bf[4];
#pragma unroll
      for (int dt = 0; dt < 4; ++dt)
        bf[dt] = frag_f16(&Vs[cur][0] + dt * 16 * 72 + kk * 32, 72, lane);
#pragma unroll
      for (int dt = 0; dt < 4; ++dt)
        o[dt] = wmma16(a, bf[dt], o[dt]);
    }
    __syncthreads();                     // done reading buf[cur] before reuse
  }

  // Normalize and write (b, L, DIM) f16
#pragma unroll
  for (int dt = 0; dt < 4; ++dt)
#pragma unroll
    for (int r = 0; r < 8; ++r) {
      int m = qb * 64 + wid * 16 + r + hi;
      float val = o[dt][r] / li[r];
      ao[((size_t)b * L_ + m) * DIM_ + h * DH + dt * 16 + (lane & 15)] = (_Float16)val;
    }
}

// ---------------------------------------------------------------------------
// Kernel 3: out = attn(f16) @ w_proj, fp32 out. BM=64, BN=256, pipelined.
// ---------------------------------------------------------------------------
__global__ __launch_bounds__(256) void proj_kernel(
    const _Float16* __restrict__ a16, const float* __restrict__ w,
    float* __restrict__ out) {
  __shared__ _Float16 Bs[256 * 40];
  const int lane = threadIdx.x & 31;
  const int wid  = threadIdx.x >> 5;
  const int wm = wid & 1, wn = wid >> 1;
  const int bn = blockIdx.x, bm = blockIdx.y;
  const int row0 = bm * 64 + wm * 32;

  v8f z = {0.f,0.f,0.f,0.f,0.f,0.f,0.f,0.f};
  v8f acc[2][4];
#pragma unroll
  for (int ms = 0; ms < 2; ++ms)
#pragma unroll
    for (int nt = 0; nt < 4; ++nt) acc[ms][nt] = z;

  v4f tmp[8];
#pragma unroll
  for (int i = 0; i < 8; ++i) {
    int f4 = threadIdx.x + 256 * i;
    int k = f4 >> 6, n4 = f4 & 63;
    tmp[i] = *(const v4f*)&w[(size_t)k * DIM_ + bn * 256 + n4 * 4];
  }

  for (int kb = 0; kb < DIM_; kb += 32) {
    __syncthreads();
#pragma unroll
    for (int i = 0; i < 8; ++i) {
      int f4 = threadIdx.x + 256 * i;
      int k = f4 >> 6, n = (f4 & 63) * 4;
      Bs[(n + 0) * 40 + k] = (_Float16)tmp[i][0];
      Bs[(n + 1) * 40 + k] = (_Float16)tmp[i][1];
      Bs[(n + 2) * 40 + k] = (_Float16)tmp[i][2];
      Bs[(n + 3) * 40 + k] = (_Float16)tmp[i][3];
    }
    __syncthreads();
    if (kb + 32 < DIM_) {
#pragma unroll
      for (int i = 0; i < 8; ++i) {
        int f4 = threadIdx.x + 256 * i;
        int k = f4 >> 6, n4 = f4 & 63;
        tmp[i] = *(const v4f*)&w[(size_t)(kb + 32 + k) * DIM_ + bn * 256 + n4 * 4];
      }
    }
    v16h a0 = frag_f16(a16 + (size_t)row0 * DIM_ + kb, DIM_, lane);
    v16h a1 = frag_f16(a16 + (size_t)(row0 + 16) * DIM_ + kb, DIM_, lane);
    v16h bf[4];
#pragma unroll
    for (int nt = 0; nt < 4; ++nt)
      bf[nt] = frag_f16(Bs + (wn * 64 + nt * 16) * 40, 40, lane);
#pragma unroll
    for (int nt = 0; nt < 4; ++nt) {
      acc[0][nt] = wmma16(a0, bf[nt], acc[0][nt]);
      acc[1][nt] = wmma16(a1, bf[nt], acc[1][nt]);
    }
  }

  const int hi = (lane & 16) ? 8 : 0;
#pragma unroll
  for (int ms = 0; ms < 2; ++ms)
#pragma unroll
    for (int nt = 0; nt < 4; ++nt) {
      int n = bn * 256 + wn * 64 + nt * 16 + (lane & 15);
#pragma unroll
      for (int r = 0; r < 8; ++r) {
        int m = row0 + ms * 16 + r + hi;
        out[(size_t)m * DIM_ + n] = acc[ms][nt][r];
      }
    }
}

// ---------------------------------------------------------------------------
extern "C" void kernel_launch(void* const* d_in, const int* in_sizes, int n_in,
                              void* d_out, int out_size, void* d_ws, size_t ws_size,
                              hipStream_t stream) {
  (void)in_sizes; (void)n_in; (void)out_size; (void)ws_size;
  const float* x      = (const float*)d_in[0];
  const float* w_qkv  = (const float*)d_in[1];
  const float* w_proj = (const float*)d_in[2];

  char* ws = (char*)d_ws;
  _Float16* qw = (_Float16*)(ws);
  _Float16* kw = (_Float16*)(ws + (size_t)16 * 1024 * 1024);
  _Float16* vw = (_Float16*)(ws + (size_t)32 * 1024 * 1024);
  _Float16* ao = (_Float16*)(ws + (size_t)48 * 1024 * 1024);
  float* out = (float*)d_out;

  qkv_rope_kernel<<<dim3(N3 / 256, (B_ * L_) / 64), 256, 0, stream>>>(x, w_qkv, qw, kw, vw);
  attn_kernel<<<dim3(L_ / 64, NH, B_), 128, 0, stream>>>(qw, kw, vw, ao);
  proj_kernel<<<dim3(DIM_ / 256, (B_ * L_) / 64), 256, 0, stream>>>(ao, w_proj, out);
}
